// muP_AttentionMechanism_35210141892840
// MI455X (gfx1250) — matrix operand order repaired and checked by reading, not compiled, not run on hardware
//
#include <hip/hip_runtime.h>
#include <hip/hip_bf16.h>
#include <cstdint>

// Flash-attention forward for B=2,H=16,L=2048,D=64, causal, scale=1/D (muP).
// bf16 WMMA (v_wmma_f32_16x16x32_bf16) for QK^T, PV, and row-sums (P x ones),
// fp32 accumulation, online softmax in exp2 domain (scale folded into Q).
// Raw v_exp_f32 (flush-to-zero tails) and v_perm_b32 bf16 packing keep the
// VALU pipe light so the matrix pipe is the limiter.
// One wave = 16 query rows; 8 waves / 256-thread block.

typedef __bf16 v16bf __attribute__((ext_vector_type(16)));
typedef float  v8f   __attribute__((ext_vector_type(8)));

#define WAVES 8
#define BM    (WAVES * 16)   // 128 query rows per block
#define NB    64             // key block size
#define DHEAD 64
#define LK    72             // LDS pitch (elements) for K / V^T tiles
#define LP    72             // LDS pitch for P tiles
#define LOG2E 1.44269504088896340736f

// exp2 with raw v_exp_f32 semantics: args are <= 0 here and tails below
// -126 must become 0 anyway, so no denormal-range guard is needed.
static __device__ __forceinline__ float fast_exp2(float x) {
#if __has_builtin(__builtin_amdgcn_exp2f)
  return __builtin_amdgcn_exp2f(x);
#else
  return exp2f(x);
#endif
}

// pack two fp32 -> two bf16 in one v_perm_b32 (truncating round)
static __device__ __forceinline__ uint32_t pk2bf(float lo, float hi) {
#if __has_builtin(__builtin_amdgcn_perm)
  return __builtin_amdgcn_perm(__float_as_uint(hi), __float_as_uint(lo),
                               0x07060302u);
#else
  return (__float_as_uint(hi) & 0xFFFF0000u) | (__float_as_uint(lo) >> 16);
#endif
}
static __device__ __forceinline__ uint16_t f2bf(float f) {
  return (uint16_t)(__float_as_uint(f) >> 16);   // truncate
}

union AB16 { v16bf v; uint32_t u[8]; };
union F8   { v8f v; float f[8]; };

__global__ __launch_bounds__(256)
void fa_fwd(const float* __restrict__ Q, const float* __restrict__ K,
            const float* __restrict__ V, float* __restrict__ O,
            int L, float logscale) {
  __shared__ uint16_t sK[NB * LK];          // sK[key][d]      (bf16 bits)
  __shared__ uint16_t sV[DHEAD * LK];       // sV[d][key]      (transposed)
  __shared__ uint16_t sP[WAVES * 16 * LP];  // per-wave P tile [m][key]

  const int tid  = threadIdx.x;
  const int wave = tid >> 5;
  const int lane = tid & 31;
  const int n    = lane & 15;   // column index inside a 16x16 tile
  const int g    = lane >> 4;   // lane half-group

  const int bh   = blockIdx.y;
  const int row0 = blockIdx.x * BM;
  const int qrow = row0 + wave * 16;        // first query row of this wave

  const float* Qb = Q + ((size_t)bh * L) * DHEAD;
  const float* Kb = K + ((size_t)bh * L) * DHEAD;
  const float* Vb = V + ((size_t)bh * L) * DHEAD;
  float*       Ob = O + ((size_t)bh * L) * DHEAD;

  // ---- Q fragments (A layout), pre-scaled by logscale so S is in exp2 domain ----
  AB16 qf[2];
  {
    const float* qp = Qb + (size_t)(qrow + n) * DHEAD;
    #pragma unroll
    for (int f = 0; f < 2; ++f) {
      const float* p0 = qp + f * 32 + 8 * g;         // v0..3 <- k = f*32+8g..
      const float* p1 = qp + f * 32 + 16 + 8 * g;    // v4..7 <- k = f*32+16+8g..
      float4 a = *(const float4*)p0;
      float4 b = *(const float4*)(p0 + 4);
      float4 c = *(const float4*)p1;
      float4 d = *(const float4*)(p1 + 4);
      qf[f].u[0] = pk2bf(a.x * logscale, a.y * logscale);
      qf[f].u[1] = pk2bf(a.z * logscale, a.w * logscale);
      qf[f].u[2] = pk2bf(b.x * logscale, b.y * logscale);
      qf[f].u[3] = pk2bf(b.z * logscale, b.w * logscale);
      qf[f].u[4] = pk2bf(c.x * logscale, c.y * logscale);
      qf[f].u[5] = pk2bf(c.z * logscale, c.w * logscale);
      qf[f].u[6] = pk2bf(d.x * logscale, d.y * logscale);
      qf[f].u[7] = pk2bf(d.z * logscale, d.w * logscale);
    }
  }

  // all-ones B fragment (for WMMA row sums of P)
  AB16 ones;
  #pragma unroll
  for (int w = 0; w < 8; ++w) ones.u[w] = 0x3F803F80u;  // bf16 1.0 | 1.0

  F8 acc[4];
  float mt[8], lt[8];
  #pragma unroll
  for (int t = 0; t < 4; ++t)
    #pragma unroll
    for (int r = 0; r < 8; ++r) acc[t].f[r] = 0.f;
  #pragma unroll
  for (int r = 0; r < 8; ++r) { mt[r] = -1e30f; lt[r] = 0.f; }

  const int li = tid >> 2;          // 0..63: K/V tile row this thread stages
  const int d0 = (tid & 3) * 16;    // 16-wide column slice

  const int kb_end = (row0 + BM) / NB;
  for (int kb = 0; kb < kb_end; ++kb) {
    __syncthreads();
    // ---- cooperative stage of K and V^T tiles into LDS (fp32 -> bf16) ----
    {
      const float* gk = Kb + (size_t)(kb * NB + li) * DHEAD + d0;
      const float* gv = Vb + (size_t)(kb * NB + li) * DHEAD + d0;
      float4 k0 = *(const float4*)(gk);
      float4 k1 = *(const float4*)(gk + 4);
      float4 k2 = *(const float4*)(gk + 8);
      float4 k3 = *(const float4*)(gk + 12);
      uint32_t* dst = (uint32_t*)&sK[li * LK + d0];
      dst[0] = pk2bf(k0.x, k0.y); dst[1] = pk2bf(k0.z, k0.w);
      dst[2] = pk2bf(k1.x, k1.y); dst[3] = pk2bf(k1.z, k1.w);
      dst[4] = pk2bf(k2.x, k2.y); dst[5] = pk2bf(k2.z, k2.w);
      dst[6] = pk2bf(k3.x, k3.y); dst[7] = pk2bf(k3.z, k3.w);
      float4 v0 = *(const float4*)(gv);
      float4 v1 = *(const float4*)(gv + 4);
      float4 v2 = *(const float4*)(gv + 8);
      float4 v3 = *(const float4*)(gv + 12);
      float vv[16] = {v0.x,v0.y,v0.z,v0.w, v1.x,v1.y,v1.z,v1.w,
                      v2.x,v2.y,v2.z,v2.w, v3.x,v3.y,v3.z,v3.w};
      #pragma unroll
      for (int j = 0; j < 16; ++j) sV[(d0 + j) * LK + li] = f2bf(vv[j]);
      if (kb + 1 < kb_end) {   // global_prefetch_b8 for the next tiles
        __builtin_prefetch(gk + (size_t)NB * DHEAD, 0, 0);
        __builtin_prefetch(gv + (size_t)NB * DHEAD, 0, 0);
      }
    }
    __syncthreads();

    if (kb * NB > qrow + 15) continue;   // this wave is fully above the diagonal

    // ---- S = Q K^T : 4 n-tiles x 2 k-fragments (already in exp2 domain) ----
    F8 S[4];
    #pragma unroll
    for (int t = 0; t < 4; ++t) {
      #pragma unroll
      for (int r = 0; r < 8; ++r) S[t].f[r] = 0.f;
      #pragma unroll
      for (int kf = 0; kf < 2; ++kf) {
        AB16 bf;
        const uint32_t* p = (const uint32_t*)&sK[(t * 16 + n) * LK + kf * 32 + g * 8];
        #pragma unroll
        for (int w = 0; w < 8; ++w) bf.u[w] = p[(w & 3) + (w >> 2) * 8];
        S[t].v = __builtin_amdgcn_wmma_f32_16x16x32_bf16(
            false, qf[kf].v, false, bf.v, (short)0, S[t].v, false, false);
      }
    }

    // ---- causal mask: wave-uniform scalar branch, off the common path ----
    if (kb * NB + NB - 1 > qrow) {
      #pragma unroll
      for (int t = 0; t < 4; ++t) {
        const int col = kb * NB + t * 16 + n;
        #pragma unroll
        for (int r = 0; r < 8; ++r) {
          const int row = qrow + r + 8 * g;
          if (col > row) S[t].f[r] = -1e30f;
        }
      }
    }

    // ---- online softmax: row max over n (16-lane shfl groups) ----
    float nm[8], alpha[8];
    #pragma unroll
    for (int r = 0; r < 8; ++r) {
      float v = fmaxf(fmaxf(S[0].f[r], S[1].f[r]), fmaxf(S[2].f[r], S[3].f[r]));
      v = fmaxf(v, __shfl_xor(v, 1, 16));
      v = fmaxf(v, __shfl_xor(v, 2, 16));
      v = fmaxf(v, __shfl_xor(v, 4, 16));
      v = fmaxf(v, __shfl_xor(v, 8, 16));
      nm[r] = fmaxf(mt[r], v);
      alpha[r] = fast_exp2(mt[r] - nm[r]);
      mt[r] = nm[r];
    }

    // ---- P = exp2(S - m) -> bf16 -> LDS (C-layout -> A-layout transpose) ----
    uint16_t* pw = &sP[(wave * 16) * LP + n];
    #pragma unroll
    for (int r = 0; r < 8; ++r) {
      float p0 = fast_exp2(S[0].f[r] - nm[r]);
      float p1 = fast_exp2(S[1].f[r] - nm[r]);
      float p2 = fast_exp2(S[2].f[r] - nm[r]);
      float p3 = fast_exp2(S[3].f[r] - nm[r]);
      uint32_t u01 = pk2bf(p0, p1);
      uint32_t u23 = pk2bf(p2, p3);
      uint16_t* row = pw + (r + 8 * g) * LP;
      row[0]  = (uint16_t)u01;
      row[16] = (uint16_t)(u01 >> 16);
      row[32] = (uint16_t)u23;
      row[48] = (uint16_t)(u23 >> 16);
    }

    // rescale O accumulator
    #pragma unroll
    for (int r = 0; r < 8; ++r)
      #pragma unroll
      for (int t = 0; t < 4; ++t) acc[t].f[r] *= alpha[r];

    // ---- reload P as A fragments ----
    AB16 pf[2];
    {
      const uint16_t* pr = &sP[(wave * 16 + n) * LP];
      #pragma unroll
      for (int f = 0; f < 2; ++f) {
        const uint32_t* c0 = (const uint32_t*)(pr + f * 32 + 8 * g);
        const uint32_t* c1 = (const uint32_t*)(pr + f * 32 + 16 + 8 * g);
        pf[f].u[0] = c0[0]; pf[f].u[1] = c0[1];
        pf[f].u[2] = c0[2]; pf[f].u[3] = c0[3];
        pf[f].u[4] = c1[0]; pf[f].u[5] = c1[1];
        pf[f].u[6] = c1[2]; pf[f].u[7] = c1[3];
      }
    }

    // ---- row sums on the matrix pipe: ls = P x ones (broadcast across lanes) ----
    F8 ls;
    #pragma unroll
    for (int r = 0; r < 8; ++r) ls.f[r] = 0.f;
    ls.v = __builtin_amdgcn_wmma_f32_16x16x32_bf16(
        false, pf[0].v, false, ones.v, (short)0, ls.v, false, false);
    ls.v = __builtin_amdgcn_wmma_f32_16x16x32_bf16(
        false, pf[1].v, false, ones.v, (short)0, ls.v, false, false);
    #pragma unroll
    for (int r = 0; r < 8; ++r) lt[r] = lt[r] * alpha[r] + ls.f[r];

    // ---- O += P V ----
    #pragma unroll
    for (int t = 0; t < 4; ++t) {
      #pragma unroll
      for (int kf = 0; kf < 2; ++kf) {
        AB16 vf;
        const uint32_t* p = (const uint32_t*)&sV[(t * 16 + n) * LK + kf * 32 + g * 8];
        #pragma unroll
        for (int w = 0; w < 8; ++w) vf.u[w] = p[(w & 3) + (w >> 2) * 8];
        acc[t].v = __builtin_amdgcn_wmma_f32_16x16x32_bf16(
            false, pf[kf].v, false, vf.v, (short)0, acc[t].v, false, false);
      }
    }
  }

  // ---- epilogue: O / l, fp32 store ----
  #pragma unroll
  for (int r = 0; r < 8; ++r) {
    float inv = 1.0f / lt[r];
    int row = qrow + r + 8 * g;
    float* op = Ob + (size_t)row * DHEAD;
    #pragma unroll
    for (int t = 0; t < 4; ++t) op[t * 16 + n] = acc[t].f[r] * inv;
  }
}

extern "C" void kernel_launch(void* const* d_in, const int* in_sizes, int n_in,
                              void* d_out, int out_size, void* d_ws, size_t ws_size,
                              hipStream_t stream) {
  (void)in_sizes; (void)n_in; (void)out_size; (void)d_ws; (void)ws_size;
  const float* Q = (const float*)d_in[0];
  const float* K = (const float*)d_in[1];
  const float* V = (const float*)d_in[2];
  float* out = (float*)d_out;
  const int L  = 2048;
  const int BH = 2 * 16;
  dim3 grid(L / BM, BH);
  const float logscale = (1.0f / (float)DHEAD) * LOG2E;  // 1/dhead in exp2 domain
  fa_fwd<<<grid, 256, 0, stream>>>(Q, K, V, out, L, logscale);
}
